// RNN_41953240547394
// MI455X (gfx1250) — matrix operand order repaired and checked by reading, not compiled, not run on hardware
//
#include <hip/hip_runtime.h>
#include <cmath>

#define INPUT  128
#define HIDDEN 256
#define OUTPUT 128
#define BATCH  64
#define SEQLEN 2048

typedef _Float16 half8  __attribute__((ext_vector_type(8)));
typedef _Float16 half16 __attribute__((ext_vector_type(16)));
typedef float    float8 __attribute__((ext_vector_type(8)));

// ---- dynamic LDS layout (halfs unless noted), all row pitches 16B-aligned ----
constexpr int WC_PITCH   = 392;                    // [W_hh | W_ih] rows: 384 used + pad
constexpr int H_PITCH    = 264;                    // h rows: 256 used + pad
constexpr int X_PITCH    = 136;                    // x_t rows: 128 used + pad
constexpr int WC_HALFS   = HIDDEN * WC_PITCH;      // 100352
constexpr int HBUF_HALFS = BATCH  * H_PITCH;       // 16896
constexpr int XBUF_HALFS = BATCH  * X_PITCH;       // 8704
constexpr size_t LDS_BYTES =
    (size_t)WC_HALFS * 2 + 2 * (size_t)HBUF_HALFS * 2 + 2 * (size_t)XBUF_HALFS * 2 + 256 * 4;
// = 200704 + 67584 + 34816 + 1024 = 304128 bytes  (< 320KB WGP limit)

union Frag16 { half16 v; half8 p[2]; };

// gfx1250 has a hardware V_TANH_F32 transcendental; use it if the builtin is
// declared, else a 2-trans-op formulation (exp2 + rcp). Both are 1-2 cycles of
// TRANS instead of the ~30-op ocml polynomial on the serial critical path.
__device__ __forceinline__ float fast_tanh(float x) {
#if __has_builtin(__builtin_amdgcn_tanhf)
    return __builtin_amdgcn_tanhf(x);
#elif __has_builtin(__builtin_amdgcn_tanh_f32)
    return __builtin_amdgcn_tanh_f32(x);
#else
    // tanh(x) = 1 - 2/(exp2(2*log2e*x) + 1)
    float e = __builtin_amdgcn_exp2f(x * 2.8853900817779268f);
    return 1.0f - 2.0f * __builtin_amdgcn_rcpf(e + 1.0f);
#endif
}

__global__ __launch_bounds__(1024, 1)
void rnn_fused_kernel(const float* __restrict__ x,
                      const float* __restrict__ W_ih,
                      const float* __restrict__ W_hh,
                      const float* __restrict__ b_ih,
                      const float* __restrict__ b_hh,
                      const float* __restrict__ W_fc,
                      const float* __restrict__ b_fc,
                      float* __restrict__ out)
{
    extern __shared__ char smem[];
    _Float16* Wc   = (_Float16*)smem;                 // [256][WC_PITCH]  f16
    _Float16* hb   = Wc + WC_HALFS;                   // 2 x [64][H_PITCH] f16
    _Float16* xb   = hb + 2 * HBUF_HALFS;             // 2 x [64][X_PITCH] f16
    float*    bias = (float*)(xb + 2 * XBUF_HALFS);   // [256] f32

    const int tid  = threadIdx.x;
    const int lane = tid & 31;
    const int wave = tid >> 5;
    const int hi   = lane >> 4;    // half-wave id (0/1)
    const int l16  = lane & 15;

    // ---------------- prologue: pack weights + zero h0 + stage x_0 ----------------
    for (int e = tid; e < WC_HALFS; e += 1024) {
        int n = e / WC_PITCH, k = e - n * WC_PITCH;
        float v = 0.f;
        if (k < HIDDEN)                 v = W_hh[n * HIDDEN + k];
        else if (k < HIDDEN + INPUT)    v = W_ih[n * INPUT + (k - HIDDEN)];
        Wc[e] = (_Float16)v;
    }
    for (int e = tid; e < 2 * HBUF_HALFS; e += 1024) hb[e] = (_Float16)0.f;
    if (tid < HIDDEN) bias[tid] = b_ih[tid] + b_hh[tid];

    // per-thread x staging geometry: 8 contiguous f32 of the 64x128 x_t slice
    const int xm = (tid * 8) >> 7;          // batch row
    const int xk = (tid * 8) & 127;         // input col
    {   // stage x at t=0
        const float* src = x + ((size_t)xm * SEQLEN + 0) * INPUT + xk;
        float4 a0 = *(const float4*)(src);
        float4 a1 = *(const float4*)(src + 4);
        half8 h8;
        h8[0]=(_Float16)a0.x; h8[1]=(_Float16)a0.y; h8[2]=(_Float16)a0.z; h8[3]=(_Float16)a0.w;
        h8[4]=(_Float16)a1.x; h8[5]=(_Float16)a1.y; h8[6]=(_Float16)a1.z; h8[7]=(_Float16)a1.w;
        *(half8*)&xb[xm * X_PITCH + xk] = h8;
    }
    __syncthreads();

    // ---------------- recurrence: 2048 dependent steps on one WGP ----------------
    // wave -> 2 adjacent-N 16x16 output tiles sharing the A fragment
    const int m_base = (wave >> 3) * 16;            // 4 row blocks
    const int n0     = ((2 * wave) & 15) * 16;      // 16 col blocks
    const int n1     = n0 + 16;
    const float bias0 = bias[n0 + l16];
    const float bias1 = bias[n1 + l16];
    const int   ar    = m_base + l16;               // A row for this lane

    int cur = 0;
#pragma unroll 1
    for (int t = 0; t < SEQLEN; ++t) {
        const _Float16* hc = hb + cur * HBUF_HALFS;
        const _Float16* xc = xb + cur * XBUF_HALFS;
        _Float16* hn = hb + (cur ^ 1) * HBUF_HALFS;
        _Float16* xn = xb + (cur ^ 1) * XBUF_HALFS;

        // issue x_{t+1} global loads EARLY: latency hides under the 24 WMMAs
        const int tn = (t + 1 < SEQLEN) ? (t + 1) : (SEQLEN - 1);
        const float* xsrc = x + ((size_t)xm * SEQLEN + tn) * INPUT + xk;
        float4 xv0 = *(const float4*)(xsrc);
        float4 xv1 = *(const float4*)(xsrc + 4);
        const int tp = (t + 2 < SEQLEN) ? (t + 2) : (SEQLEN - 1);
        __builtin_prefetch(x + ((size_t)xm * SEQLEN + tp) * INPUT + xk, 0, 1);

        float8 c0, c1;
#pragma unroll
        for (int j = 0; j < 8; ++j) { c0[j] = bias0; c1[j] = bias1; }

        // K = 384 = 256 (h @ W_hh^T) + 128 (x_t @ W_ih^T), 12 WMMA K-blocks
#pragma unroll
        for (int kk = 0; kk < 12; ++kk) {
            const int kb = kk * 32;
            Frag16 a;
            if (kk < 8) {
                const _Float16* ap = hc + ar * H_PITCH + kb + 8 * hi;
                a.p[0] = *(const half8*)(ap);
                a.p[1] = *(const half8*)(ap + 16);
            } else {
                const _Float16* ap = xc + ar * X_PITCH + (kb - 256) + 8 * hi;
                a.p[0] = *(const half8*)(ap);
                a.p[1] = *(const half8*)(ap + 16);
            }
            Frag16 bf0, bf1;
            const _Float16* bp0 = Wc + (n0 + l16) * WC_PITCH + kb + 8 * hi;
            bf0.p[0] = *(const half8*)(bp0);
            bf0.p[1] = *(const half8*)(bp0 + 16);
            const _Float16* bp1 = Wc + (n1 + l16) * WC_PITCH + kb + 8 * hi;
            bf1.p[0] = *(const half8*)(bp1);
            bf1.p[1] = *(const half8*)(bp1 + 16);
            c0 = __builtin_amdgcn_wmma_f32_16x16x32_f16(false, a.v, false, bf0.v,
                                                        (short)0, c0, false, false);
            c1 = __builtin_amdgcn_wmma_f32_16x16x32_f16(false, a.v, false, bf1.v,
                                                        (short)0, c1, false, false);
        }

        // store staged x_{t+1} (loads were issued before the K-loop)
        {
            half8 h8;
            h8[0]=(_Float16)xv0.x; h8[1]=(_Float16)xv0.y; h8[2]=(_Float16)xv0.z; h8[3]=(_Float16)xv0.w;
            h8[4]=(_Float16)xv1.x; h8[5]=(_Float16)xv1.y; h8[6]=(_Float16)xv1.z; h8[7]=(_Float16)xv1.w;
            *(half8*)&xn[xm * X_PITCH + xk] = h8;
        }

        // h_{t+1} = tanh(pre): all 16 tanh first (independent trans ops pipeline),
        // then cvt+store f16 into next buffer
        float t0[8], t1[8];
#pragma unroll
        for (int v = 0; v < 8; ++v) { t0[v] = fast_tanh(c0[v]); t1[v] = fast_tanh(c1[v]); }
#pragma unroll
        for (int v = 0; v < 8; ++v) {
            int m = m_base + v + 8 * hi;   // D layout: VGPR v -> M = v + 8*(lane>=16)
            hn[m * H_PITCH + n0 + l16] = (_Float16)t0[v];
            hn[m * H_PITCH + n1 + l16] = (_Float16)t1[v];
        }
        __syncthreads();
        cur ^= 1;
    }

    // ---------------- epilogue: out = h_T @ W_fc^T + b_fc ----------------
    // repack W_fc (f16) into the Wc region, pitch H_PITCH
    for (int e = tid; e < OUTPUT * H_PITCH; e += 1024) {
        int n = e / H_PITCH, k = e - n * H_PITCH;
        Wc[e] = (_Float16)((k < HIDDEN) ? W_fc[n * HIDDEN + k] : 0.f);
    }
    __syncthreads();

    const _Float16* hf = hb + cur * HBUF_HALFS;
    const int fm = (wave >> 3) * 16;    // 4 row blocks
    const int fn = (wave & 7) * 16;     // 8 col blocks -> 32 tiles, one per wave
    const int fr = fm + l16;
    const float cb = b_fc[fn + l16];
    float8 c;
#pragma unroll
    for (int j = 0; j < 8; ++j) c[j] = cb;
#pragma unroll
    for (int kk = 0; kk < 8; ++kk) {
        const int kb = kk * 32;
        Frag16 a, b;
        const _Float16* ap = hf + fr * H_PITCH + kb + 8 * hi;
        a.p[0] = *(const half8*)(ap);
        a.p[1] = *(const half8*)(ap + 16);
        const _Float16* bp = Wc + (fn + l16) * H_PITCH + kb + 8 * hi;
        b.p[0] = *(const half8*)(bp);
        b.p[1] = *(const half8*)(bp + 16);
        c = __builtin_amdgcn_wmma_f32_16x16x32_f16(false, a.v, false, b.v,
                                                   (short)0, c, false, false);
    }
#pragma unroll
    for (int v = 0; v < 8; ++v) {
        int m = fm + v + 8 * hi;
        out[m * OUTPUT + fn + l16] = c[v];
    }
}

extern "C" void kernel_launch(void* const* d_in, const int* in_sizes, int n_in,
                              void* d_out, int out_size, void* d_ws, size_t ws_size,
                              hipStream_t stream) {
    const float* x    = (const float*)d_in[0];
    const float* W_ih = (const float*)d_in[1];
    const float* W_hh = (const float*)d_in[2];
    const float* b_ih = (const float*)d_in[3];
    const float* b_hh = (const float*)d_in[4];
    const float* W_fc = (const float*)d_in[5];
    const float* b_fc = (const float*)d_in[6];

    // allow >64KB dynamic LDS (CDNA5 WGP: 320KB/workgroup); deterministic, idempotent
    (void)hipFuncSetAttribute((const void*)rnn_fused_kernel,
                              hipFuncAttributeMaxDynamicSharedMemorySize,
                              (int)LDS_BYTES);

    rnn_fused_kernel<<<1, 1024, (int)LDS_BYTES, stream>>>(
        x, W_ih, W_hh, b_ih, b_hh, W_fc, b_fc, (float*)d_out);
}
